// TensorProductLinearGate_47107201302529
// MI455X (gfx1250) — matrix operand order repaired and checked by reading, not compiled, not run on hardware
//
#include <hip/hip_runtime.h>

// ---------------------------------------------------------------------------
// TensorProductLinearGate for MI455X (gfx1250, wave32, WMMA f16->f32 accum)
//
// Factored form:
//   G1  = x0    @ Wcat0  (Wcat0 = [W0[:128,:]/16 | W1o[:128,:]/16]          -> 512 cols)
//   G2i = x1_i  @ Wcat1  (Wcat1 = [W0[128:,:]/(16*sqrt3) | W1o[128:,:]/16 | W1e/16] -> 640)
//   s       = y0*G1[:,:384] + sum_i y1_i*G2i[:,:384] + b0
//   v1o_i   = (y1_i*G1[:,384+o] + y0*G2i[:,384+o]) * sigmoid(s[:,o])
//   v1e_i   = (y1_k*G2j[:,512+o] - y1_j*G2k[:,512+o]) * sigmoid(s[:,128+o])
//   feats   = silu(s[:,256:384])
// All 1/16 etc. scales folded into f16 weights staged once into d_ws.
// ---------------------------------------------------------------------------

typedef _Float16 h16;
typedef __attribute__((ext_vector_type(16))) _Float16 v16h;
typedef __attribute__((ext_vector_type(8)))  _Float16 v8h;
typedef __attribute__((ext_vector_type(8)))  float    v8f;

#define OUT_STRIDE 896  // 128 feats + 384 v1o + 384 v1e

// ---- weight prep: f32 -> f16, transposed (row = output col, 128 K contig), scaled
__global__ __launch_bounds__(256) void prep_weights_kernel(
    const float* __restrict__ W0, const float* __restrict__ W1o,
    const float* __restrict__ W1e, h16* __restrict__ wt)
{
  const float s16  = 1.0f / 16.0f;                           // 1/sqrt(2*MUL)
  const float s163 = 1.0f / (16.0f * 1.7320508075688772f);   // also fold 1/sqrt(3)
  const int n0 = 512 * 128;                                  // Wcat0^T  [512][128]
  const int n1 = 640 * 128;                                  // Wcat1^T  [640][128]
  int i = blockIdx.x * 256 + threadIdx.x;
  if (i < n0) {
    int o = i >> 7, c = i & 127;
    float v = (o < 384) ? W0[c * 384 + o] * s16
                        : W1o[c * 128 + (o - 384)] * s16;
    wt[i] = (h16)v;
  } else if (i < n0 + n1) {
    int i1 = i - n0;
    int o = i1 >> 7, c = i1 & 127;
    float v;
    if (o < 384)      v = W0[(128 + c) * 384 + o] * s163;
    else if (o < 512) v = W1o[(128 + c) * 128 + (o - 384)] * s16;
    else              v = W1e[c * 128 + (o - 512)] * s16;   // 1/(sqrt2*sqrt(MUL)) == 1/16
    wt[n0 + i1] = (h16)v;
  }
}

__device__ __forceinline__ v8f wmma_f16(v16h a, v16h b, v8f c) {
  return __builtin_amdgcn_wmma_f32_16x16x32_f16(false, a, false, b, (short)0, c,
                                                false, false);
}

// A fragment (16x32 f16) from LDS row-major [16][128] tile.
// lanes 0-15: M=lane, halves = K[kb..kb+8) then K[kb+16..kb+24), kb = ks*32
// lanes 16-31: same with kb += 8   (per CDNA5 ISA 16-bit A layout)
__device__ __forceinline__ v16h load_a_lds(const h16* base, int lane, int ks) {
  int m  = lane & 15;
  int kb = ks * 32 + (lane >> 4) * 8;
  const h16* r = base + m * 128 + kb;
  v8h lo = *(const v8h*)(r);
  v8h hi = *(const v8h*)(r + 16);
  v16h a;
#pragma unroll
  for (int i = 0; i < 8; ++i) { a[i] = lo[i]; a[8 + i] = hi[i]; }
  return a;
}

// B fragment (32x16 f16): lane n holds column n, same K order as A:
// lanes 0-15 -> K[kb..kb+8) then K[kb+16..kb+24), lanes 16-31 with kb += 8.
// Weights stored transposed => two contiguous 16B loads per lane.
__device__ __forceinline__ v16h load_b_glb(const h16* wt, int rowbase, int lane, int ks) {
  size_t r = (size_t)(rowbase + (lane & 15));
  const h16* p = wt + r * 128 + ks * 32 + (lane >> 4) * 8;
  v8h lo = *(const v8h*)(p);
  v8h hi = *(const v8h*)(p + 16);
  v16h b;
#pragma unroll
  for (int i = 0; i < 8; ++i) { b[i] = lo[i]; b[8 + i] = hi[i]; }
  return b;
}

__global__ __launch_bounds__(256) void tplg_kernel(
    const float* __restrict__ x0, const float* __restrict__ x1,
    const float* __restrict__ y0, const float* __restrict__ y1,
    const float* __restrict__ b0, const h16* __restrict__ wt,
    float* __restrict__ out)
{
  __shared__ alignas(16) h16 xh[4][16][128];   // x0, x1x, x1y, x1z  (f16) 16KB
  __shared__ float yy[16][4];                  // y0, y1x, y1y, y1z
  __shared__ float gatesS[16][256];            // sigmoid(s[:, :256])  16KB

  const int t     = threadIdx.x;
  const int lane  = t & 31;
  const int wave  = t >> 5;        // 0..7
  const int node0 = blockIdx.x * 16;

  // ---- phase A: stage inputs to LDS as f16 -------------------------------
  if (t < 64) {
    int m = t >> 2, c = t & 3;
    yy[m][c] = (c == 0) ? y0[node0 + m] : y1[(size_t)(node0 + m) * 3 + (c - 1)];
  }
#pragma unroll
  for (int it = 0; it < 8; ++it) {
    int p = t + it * 256;                 // 0..2047
    int m = p >> 7, c = p & 127;
    size_t g = (size_t)(node0 + m) * 128 + c;
    xh[0][m][c] = (h16)x0[g];
    const float* xp = x1 + g * 3;
    xh[1][m][c] = (h16)xp[0];
    xh[2][m][c] = (h16)xp[1];
    xh[3][m][c] = (h16)xp[2];
  }
  __syncthreads();

  // per-lane node scalars for D-fragment rows (m = (lane>=16 ? 8:0) + v)
  const int lg = lane >> 4;
  const int ln = lane & 15;
  float ry0[8], ry1x[8], ry1y[8], ry1z[8];
#pragma unroll
  for (int v = 0; v < 8; ++v) {
    int m = lg * 8 + v;
    ry0[v]  = yy[m][0];
    ry1x[v] = yy[m][1];
    ry1y[v] = yy[m][2];
    ry1z[v] = yy[m][3];
  }

  const h16* W0T = wt;               // Wcat0^T [512][128]
  const h16* W1T = wt + 512 * 128;   // Wcat1^T [640][128]

  v8f accG[3][4];  // 3 s-tiles x {G1, G2x, G2y, G2z}
  v8f accO[7];     // {A, B0,B1,B2, H0,H1,H2} for this wave's o-tile
#pragma unroll
  for (int j = 0; j < 3; ++j)
#pragma unroll
    for (int q = 0; q < 4; ++q) accG[j][q] = (v8f){};
#pragma unroll
  for (int q = 0; q < 7; ++q) accO[q] = (v8f){};

  const int jlist[3] = {wave, wave + 8, wave + 16};  // 24 s-tiles over 8 waves
  const int ob = wave;                               // 8 o-tiles

  // ---- phase B: WMMA over K = 128 in 4 steps of 32 -----------------------
#pragma unroll
  for (int ks = 0; ks < 4; ++ks) {
    v16h a0 = load_a_lds(&xh[0][0][0], lane, ks);
    v16h a1 = load_a_lds(&xh[1][0][0], lane, ks);
    v16h a2 = load_a_lds(&xh[2][0][0], lane, ks);
    v16h a3 = load_a_lds(&xh[3][0][0], lane, ks);
#pragma unroll
    for (int jj = 0; jj < 3; ++jj) {
      int j = jlist[jj];
      v16h bc0 = load_b_glb(W0T, j * 16, lane, ks);
      v16h bc1 = load_b_glb(W1T, j * 16, lane, ks);   // reused 3x
      accG[jj][0] = wmma_f16(a0, bc0, accG[jj][0]);
      accG[jj][1] = wmma_f16(a1, bc1, accG[jj][1]);
      accG[jj][2] = wmma_f16(a2, bc1, accG[jj][2]);
      accG[jj][3] = wmma_f16(a3, bc1, accG[jj][3]);
    }
    v16h bA = load_b_glb(W0T, 384 + ob * 16, lane, ks);
    v16h bB = load_b_glb(W1T, 384 + ob * 16, lane, ks);  // reused 3x
    v16h bH = load_b_glb(W1T, 512 + ob * 16, lane, ks);  // reused 3x
    accO[0] = wmma_f16(a0, bA, accO[0]);
    accO[1] = wmma_f16(a1, bB, accO[1]);
    accO[2] = wmma_f16(a2, bB, accO[2]);
    accO[3] = wmma_f16(a3, bB, accO[3]);
    accO[4] = wmma_f16(a1, bH, accO[4]);
    accO[5] = wmma_f16(a2, bH, accO[5]);
    accO[6] = wmma_f16(a3, bH, accO[6]);
  }

  // ---- s epilogue: gates -> LDS, feats -> global -------------------------
#pragma unroll
  for (int jj = 0; jj < 3; ++jj) {
    int j = jlist[jj];
    int q = j * 16 + ln;          // absolute s column, wave-uniform region
    float bias = b0[q];
#pragma unroll
    for (int v = 0; v < 8; ++v) {
      int m = lg * 8 + v;
      float s = ry0[v] * accG[jj][0][v] + ry1x[v] * accG[jj][1][v] +
                ry1y[v] * accG[jj][2][v] + ry1z[v] * accG[jj][3][v] + bias;
      float sig = 1.0f / (1.0f + __expf(-s));
      if (q < 256) {
        gatesS[m][q] = sig;                                  // gate
      } else {
        out[(size_t)(node0 + m) * OUT_STRIDE + (q - 256)] = s * sig;  // silu feat
      }
    }
  }
  __syncthreads();

  // ---- o epilogue: gated 1o / 1e vectors ---------------------------------
  const int o = ob * 16 + ln;     // output channel 0..127
#pragma unroll
  for (int v = 0; v < 8; ++v) {
    int m = lg * 8 + v;
    float go = gatesS[m][o];
    float ge = gatesS[m][128 + o];
    float A  = accO[0][v];
    float B0 = accO[1][v], B1 = accO[2][v], B2 = accO[3][v];
    float H0 = accO[4][v], H1 = accO[5][v], H2 = accO[6][v];
    float* orow = out + (size_t)(node0 + m) * OUT_STRIDE;
    // v1o[o][i] = (y1_i*A + y0*B_i) * gate0
    orow[128 + o * 3 + 0] = (ry1x[v] * A + ry0[v] * B0) * go;
    orow[128 + o * 3 + 1] = (ry1y[v] * A + ry0[v] * B1) * go;
    orow[128 + o * 3 + 2] = (ry1z[v] * A + ry0[v] * B2) * go;
    // v1e = cross-combination of H_i, * gate1
    orow[512 + o * 3 + 0] = (ry1z[v] * H1 - ry1y[v] * H2) * ge;
    orow[512 + o * 3 + 1] = (ry1x[v] * H2 - ry1z[v] * H0) * ge;
    orow[512 + o * 3 + 2] = (ry1y[v] * H0 - ry1x[v] * H1) * ge;
  }
}

extern "C" void kernel_launch(void* const* d_in, const int* in_sizes, int n_in,
                              void* d_out, int out_size, void* d_ws, size_t ws_size,
                              hipStream_t stream) {
  const float* x0  = (const float*)d_in[0];
  const float* x1  = (const float*)d_in[1];
  const float* y0  = (const float*)d_in[2];
  const float* y1  = (const float*)d_in[3];
  const float* W0  = (const float*)d_in[4];
  const float* b0  = (const float*)d_in[5];
  const float* W1o = (const float*)d_in[6];
  const float* W1e = (const float*)d_in[7];
  float* out = (float*)d_out;
  h16* wt = (h16*)d_ws;   // needs 288 KB of workspace

  int N = in_sizes[0] / 128;   // nodes (65536)

  // 1) fold scales, transpose, convert weights to f16 (147456 elements)
  prep_weights_kernel<<<(147456 + 255) / 256, 256, 0, stream>>>(W0, W1o, W1e, wt);

  // 2) fused TP + linear + gate; one 16-node tile per workgroup (8 waves)
  tplg_kernel<<<N / 16, 256, 0, stream>>>(x0, x1, y0, y1, b0, wt, out);
}
